// Net_16028817948751
// MI455X (gfx1250) — matrix operand order repaired and checked by reading, not compiled, not run on hardware
//
#include <hip/hip_runtime.h>
#include <hip/hip_bf16.h>

typedef __attribute__((ext_vector_type(16))) _Float16 v16h;
typedef __attribute__((ext_vector_type(8)))  _Float16 v8h;
typedef __attribute__((ext_vector_type(4)))  _Float16 v4h;
typedef __attribute__((ext_vector_type(8)))  float    v8f;
typedef __attribute__((ext_vector_type(4)))  int      v4i;

#define GLOBAL_AS __attribute__((address_space(1)))
#define LOCAL_AS  __attribute__((address_space(3)))

#define H 64            // hidden / feature width (fixed by problem)
#define BANDS_C 3
#define BATCH_C 50
#define CR_C 6000                        // coef rows per graph (300000/50)
#define POOLSZ (BATCH_C * BANDS_C * H)   // 9600 floats
#define APAD 8                           // LDS row pad (16B-aligned, kills bank conflicts)

#if defined(__has_builtin)
#if __has_builtin(__builtin_amdgcn_global_load_async_to_lds_b128)
#define HAVE_ASYNC_LDS 1
#endif
#endif

// ---------------------------------------------------------------------------
// Degree / normalization
// ---------------------------------------------------------------------------
__global__ __launch_bounds__(256) void deg_init(float* __restrict__ deg, int N) {
    int i = blockIdx.x * blockDim.x + threadIdx.x;
    if (i < N) deg[i] = 1.0f;                       // self-loop contributes 1
}

__global__ __launch_bounds__(256) void deg_count(const int* __restrict__ row,
                                                 float* __restrict__ deg, int E) {
    int i = blockIdx.x * blockDim.x + threadIdx.x;
    if (i < E) atomicAdd(deg + row[i], 1.0f);
}

__global__ __launch_bounds__(256) void make_dinv(float* __restrict__ deg, int N) {
    int i = blockIdx.x * blockDim.x + threadIdx.x;
    if (i < N) deg[i] = rsqrtf(deg[i]);             // deg >= 1 always
}

// ---------------------------------------------------------------------------
// GEMM: C[M,64] = A[M,64] @ W[64,64], f16 WMMA with f32 accumulate.
// Block = 256 threads (8 waves); block covers 128 rows; each wave a 16-row
// strip across all 64 output columns (4 N-tiles x 2 K-chunks -> 8 WMMA).
// W staged TRANSPOSED so B fragments are contiguous 16B LDS chunks.
// ---------------------------------------------------------------------------
__global__ __launch_bounds__(256)
void gemm_n64_wmma(const float* __restrict__ A, const float* __restrict__ W,
                   float* __restrict__ C, int M) {
    __shared__ _Float16 sA[128][H + APAD];
    __shared__ _Float16 sWt[H][H + APAD];    // sWt[n][k] = W[k][n]

    const int tid  = threadIdx.x;
    const int wave = tid >> 5;
    const int lane = tid & 31;
    const int m0   = blockIdx.x * 128;

    // Stage A tile (128x64) as f16 (8B LDS stores).
    for (int i = tid; i < 128 * H / 4; i += 256) {
        int idx = i * 4;
        int r = idx >> 6, cc = idx & 63;
        int gr = m0 + r;
        float4 v = (gr < M) ? *(const float4*)(A + (size_t)gr * H + cc)
                            : make_float4(0.f, 0.f, 0.f, 0.f);
        v4h h4 = { (_Float16)v.x, (_Float16)v.y, (_Float16)v.z, (_Float16)v.w };
        *(v4h*)&sA[r][cc] = h4;
    }
    // Stage W transposed.
    for (int i = tid; i < H * H / 4; i += 256) {
        int idx = i * 4;
        int r = idx >> 6, cc = idx & 63;
        float4 v = *(const float4*)(W + idx);
        sWt[cc + 0][r] = (_Float16)v.x;
        sWt[cc + 1][r] = (_Float16)v.y;
        sWt[cc + 2][r] = (_Float16)v.z;
        sWt[cc + 3][r] = (_Float16)v.w;
    }
    __syncthreads();

    const int mbase = wave * 16;
    const int ml    = lane & 15;       // M index for A frag / N index for B,C
    const int hi    = lane >> 4;

    // A fragments: per lane 16 contiguous halves at k = 16*hi + j
    // (and +32 for K-chunk 1), matching the B fragment k order.
    const _Float16* ap = &sA[mbase + ml][0];
    v8h a0lo = *(const v8h*)(ap + 16 * hi);
    v8h a0hi = *(const v8h*)(ap + 16 * hi + 8);
    v8h a1lo = *(const v8h*)(ap + 32 + 16 * hi);
    v8h a1hi = *(const v8h*)(ap + 40 + 16 * hi);
    v16h af0 = __builtin_shufflevector(a0lo, a0hi, 0,1,2,3,4,5,6,7,8,9,10,11,12,13,14,15);
    v16h af1 = __builtin_shufflevector(a1lo, a1hi, 0,1,2,3,4,5,6,7,8,9,10,11,12,13,14,15);

    const bool full = (m0 + 128 <= M);   // uniform: fast unguarded store path

#pragma unroll
    for (int nt = 0; nt < 4; ++nt) {
        const int n0 = nt * 16;
        // B fragment: rows K striped across lane halves -> k = 16*hi + j;
        // transposed staging makes the 16 halves contiguous.
        const _Float16* bp = &sWt[n0 + ml][0];
        v8h b0lo = *(const v8h*)(bp + 16 * hi);
        v8h b0hi = *(const v8h*)(bp + 16 * hi + 8);
        v8h b1lo = *(const v8h*)(bp + 32 + 16 * hi);
        v8h b1hi = *(const v8h*)(bp + 40 + 16 * hi);
        v16h bf0 = __builtin_shufflevector(b0lo, b0hi, 0,1,2,3,4,5,6,7,8,9,10,11,12,13,14,15);
        v16h bf1 = __builtin_shufflevector(b1lo, b1hi, 0,1,2,3,4,5,6,7,8,9,10,11,12,13,14,15);

        v8f acc = {};
        acc = __builtin_amdgcn_wmma_f32_16x16x32_f16(false, af0, false, bf0,
                                                     (short)0, acc, false, false);
        acc = __builtin_amdgcn_wmma_f32_16x16x32_f16(false, af1, false, bf1,
                                                     (short)0, acc, false, false);

        const int rbase = m0 + mbase + hi * 8;       // C: VGPR v -> row hi*8+v
        float* cp = C + (size_t)rbase * H + n0 + ml;
        if (full) {
#pragma unroll
            for (int v = 0; v < 8; ++v) cp[v * H] = acc[v];
        } else {
#pragma unroll
            for (int v = 0; v < 8; ++v)
                if (rbase + v < M) cp[v * H] = acc[v];
        }
    }
}

// ---------------------------------------------------------------------------
// Edge aggregation: out[r] += dinv[r]*dinv[c]*h[c]  (4 work-items per edge,
// 16 features each -> 16 global f32 atomics)
// ---------------------------------------------------------------------------
__global__ __launch_bounds__(256)
void agg_edges(const int* __restrict__ row, const int* __restrict__ col,
               const float* __restrict__ dinv, const float* __restrict__ hin,
               float* __restrict__ out, int E) {
    int i = blockIdx.x * blockDim.x + threadIdx.x;
    int total = E * 4;
    if (i >= total) return;
    int e = i >> 2, q = i & 3;
    int r = row[e], c = col[e];
    float w = dinv[r] * dinv[c];
    const float4* hp = (const float4*)(hin + (size_t)c * H) + q * 4;
    float* op = out + (size_t)r * H + q * 16;
#pragma unroll
    for (int j = 0; j < 4; ++j) {
        float4 hv = hp[j];
        atomicAdd(op + j * 4 + 0, w * hv.x);
        atomicAdd(op + j * 4 + 1, w * hv.y);
        atomicAdd(op + j * 4 + 2, w * hv.z);
        atomicAdd(op + j * 4 + 3, w * hv.w);
    }
}

// io = relu(io + dinv^2 * hpre + bias)   (self-loop term + bias + relu)
__global__ __launch_bounds__(256)
void selfloop_bias_relu(float* __restrict__ io, const float* __restrict__ hpre,
                        const float* __restrict__ dinv, const float* __restrict__ bias,
                        int N) {
    int i = blockIdx.x * blockDim.x + threadIdx.x;   // N*16 float4 chunks
    if (i >= N * 16) return;
    int node = i >> 4, q = i & 15;
    float di = dinv[node];
    float w = di * di;
    float4 a = ((const float4*)io)[i];
    float4 hp = ((const float4*)hpre)[i];
    float4 b = *(const float4*)(bias + q * 4);
    float4 o;
    o.x = fmaxf(a.x + w * hp.x + b.x, 0.f);
    o.y = fmaxf(a.y + w * hp.y + b.y, 0.f);
    o.z = fmaxf(a.z + w * hp.z + b.z, 0.f);
    o.w = fmaxf(a.w + w * hp.w + b.w, 0.f);
    ((float4*)io)[i] = o;
}

// ---------------------------------------------------------------------------
// Fused framelet SpMM + per-(graph,band) sum pooling.
// pooled[seg(row)] += val * h[col]; seg depends only on row, so the [NCR,H]
// coefficient matrix is never materialized. LDS-privatized (ds_add_f32),
// sparse non-zero flush to global atomics.
// ---------------------------------------------------------------------------
__global__ __launch_bounds__(256)
void framelet_pool(const int* __restrict__ frow, const int* __restrict__ fcol,
                   const float* __restrict__ fval, const int* __restrict__ dindex,
                   const float* __restrict__ h, float* __restrict__ pooled,
                   int nz) {
    __shared__ float sp[POOLSZ];
    for (int i = threadIdx.x; i < POOLSZ; i += blockDim.x) sp[i] = 0.f;
    __syncthreads();

    int per   = (nz + gridDim.x - 1) / gridDim.x;
    int start = blockIdx.x * per;
    int end   = min(start + per, nz);

    for (int i = start * 4 + (int)threadIdx.x; i < end * 4; i += blockDim.x) {
        int e = i >> 2, q = i & 3;
        int r = frow[e], c = fcol[e];
        float v = fval[e];
        int seg = (r / CR_C) * BANDS_C + dindex[r];   // compile-time divisor
        const float4* hp = (const float4*)(h + (size_t)c * H) + q * 4;
        float* sb = sp + seg * H + q * 16;
#pragma unroll
        for (int j = 0; j < 4; ++j) {
            float4 hv = hp[j];
            atomicAdd(sb + j * 4 + 0, v * hv.x);
            atomicAdd(sb + j * 4 + 1, v * hv.y);
            atomicAdd(sb + j * 4 + 2, v * hv.z);
            atomicAdd(sb + j * 4 + 3, v * hv.w);
        }
    }
    __syncthreads();
    for (int i = threadIdx.x; i < POOLSZ; i += blockDim.x) {
        float s = sp[i];
        if (s != 0.f) atomicAdd(pooled + i, s);   // chunk touches few segs
    }
}

// ---------------------------------------------------------------------------
// MLP head: out = relu(xp @ fcW1 + fcb1) @ fcW2 + fcb2 ; xp = pooled [50,192]
// Tiny (1.3 MFLOP) -> one workgroup. Staging uses CDNA5 async global->LDS
// copies when the toolchain exposes them.
// ---------------------------------------------------------------------------
__global__ __launch_bounds__(256)
void mlp_head(const float* __restrict__ pooled,
              const float* __restrict__ fcW1, const float* __restrict__ fcb1,
              const float* __restrict__ fcW2, const float* __restrict__ fcb2,
              float* __restrict__ out) {
    __shared__ float sx[POOLSZ];                  // 9600
    __shared__ float sh[BATCH_C * H];             // 3200
#if defined(HAVE_ASYNC_LDS) && defined(__AMDGCN__)
    for (int i = threadIdx.x; i < POOLSZ / 4; i += 256) {
        GLOBAL_AS v4i* gsrc = (GLOBAL_AS v4i*)(pooled + i * 4);   // flat->AS1 cast
        LOCAL_AS  v4i* ldst = (LOCAL_AS  v4i*)(sx + i * 4);       // flat->AS3 cast
        __builtin_amdgcn_global_load_async_to_lds_b128(gsrc, ldst, 0, 0);
    }
#if __has_builtin(__builtin_amdgcn_s_wait_asynccnt)
    __builtin_amdgcn_s_wait_asynccnt(0);
#else
    asm volatile("s_wait_asynccnt 0x0" ::: "memory");
#endif
    __syncthreads();
#else
    for (int i = threadIdx.x; i < POOLSZ; i += 256) sx[i] = pooled[i];
    __syncthreads();
#endif
    for (int i = threadIdx.x; i < BATCH_C * H; i += 256) {
        int g = i >> 6, j = i & 63;
        float acc = fcb1[j];
        const float* xg = sx + g * (BANDS_C * H);
        for (int k = 0; k < BANDS_C * H; ++k) acc += xg[k] * fcW1[k * H + j];
        sh[i] = fmaxf(acc, 0.f);
    }
    __syncthreads();
    for (int i = threadIdx.x; i < BATCH_C * 10; i += 256) {
        int g = i / 10, c = i % 10;
        float acc = fcb2[c];
        const float* hg = sh + g * H;
        for (int k = 0; k < H; ++k) acc += hg[k] * fcW2[k * 10 + c];
        out[i] = acc;
    }
}

// ---------------------------------------------------------------------------
extern "C" void kernel_launch(void* const* d_in, const int* in_sizes, int n_in,
                              void* d_out, int out_size, void* d_ws, size_t ws_size,
                              hipStream_t stream) {
    const float* x      = (const float*)d_in[0];
    const int*   edge   = (const int*)  d_in[1];
    // d_in[2] = batch (unused: batch = node/n by construction)
    const int*   frow   = (const int*)  d_in[3];
    const int*   fcol   = (const int*)  d_in[4];
    const float* fval   = (const float*)d_in[5];
    const int*   dindex = (const int*)  d_in[6];
    // d_in[7] = batch_size (device scalar; B fixed at 50)
    const float* W1   = (const float*)d_in[8];
    const float* b1   = (const float*)d_in[9];
    const float* W2   = (const float*)d_in[10];
    const float* b2   = (const float*)d_in[11];
    const float* fcW1 = (const float*)d_in[12];
    const float* fcb1 = (const float*)d_in[13];
    const float* fcW2 = (const float*)d_in[14];
    const float* fcb2 = (const float*)d_in[15];
    float* out = (float*)d_out;

    const int N   = in_sizes[0] / H;        // 100000
    const int E   = in_sizes[1] / 2;        // 1600000
    const int NZ  = in_sizes[3];            // 2400000

    const int* erow = edge;
    const int* ecol = edge + E;

    // Workspace layout (float): deg/dinv[N] | tmp1[N*64] | tmp2[N*64] | pooled[9600]
    float* deg    = (float*)d_ws;
    float* tmp1   = deg + N;
    float* tmp2   = tmp1 + (size_t)N * H;
    float* pooled = tmp2 + (size_t)N * H;

    const int T = 256;
    dim3 blk(T);

    // normalization (shared by both GCN layers)
    deg_init  <<<(N + T - 1) / T, blk, 0, stream>>>(deg, N);
    deg_count <<<(E + T - 1) / T, blk, 0, stream>>>(erow, deg, E);
    make_dinv <<<(N + T - 1) / T, blk, 0, stream>>>(deg, N);

    // layer 1
    gemm_n64_wmma<<<(N + 127) / 128, blk, 0, stream>>>(x, W1, tmp1, N);
    (void)hipMemsetAsync(tmp2, 0, (size_t)N * H * sizeof(float), stream);
    agg_edges<<<(E * 4 + T - 1) / T, blk, 0, stream>>>(erow, ecol, deg, tmp1, tmp2, E);
    selfloop_bias_relu<<<(N * 16 + T - 1) / T, blk, 0, stream>>>(tmp2, tmp1, deg, b1, N);

    // layer 2
    gemm_n64_wmma<<<(N + 127) / 128, blk, 0, stream>>>(tmp2, W2, tmp1, N);
    (void)hipMemsetAsync(tmp2, 0, (size_t)N * H * sizeof(float), stream);
    agg_edges<<<(E * 4 + T - 1) / T, blk, 0, stream>>>(erow, ecol, deg, tmp1, tmp2, E);
    selfloop_bias_relu<<<(N * 16 + T - 1) / T, blk, 0, stream>>>(tmp2, tmp1, deg, b2, N);

    // fused framelet transform + pooling
    (void)hipMemsetAsync(pooled, 0, POOLSZ * sizeof(float), stream);
    framelet_pool<<<512, blk, 0, stream>>>(frow, fcol, fval, dindex, tmp2, pooled, NZ);

    // head
    mlp_head<<<1, blk, 0, stream>>>(pooled, fcW1, fcb1, fcW2, fcb2, out);
}
